// _DecoderLayer_35450660061617
// MI455X (gfx1250) — matrix operand order repaired and checked by reading, not compiled, not run on hardware
//
#include <hip/hip_runtime.h>
#include <hip/hip_bf16.h>
#include <math.h>

typedef __attribute__((ext_vector_type(16))) __bf16 v16bf;
typedef __attribute__((ext_vector_type(4)))  __bf16 v4bf;
typedef __attribute__((ext_vector_type(8)))  float  v8f;
typedef __attribute__((ext_vector_type(4)))  unsigned int uint32x4;
typedef __attribute__((ext_vector_type(8)))  int    int32x8;
typedef __attribute__((ext_vector_type(4)))  int    int32x4;

#define WMMA_BF16(a, b, c) \
  __builtin_amdgcn_wmma_f32_16x16x32_bf16(false, (a), false, (b), (short)0, (c), false, false)

#define NEGINF (-__builtin_inff())

#if defined(__gfx1250__) && __has_builtin(__builtin_amdgcn_tensor_load_to_lds) && \
    __has_builtin(__builtin_amdgcn_s_wait_tensorcnt)
#define USE_TDM 1
#else
#define USE_TDM 0
#endif

// ---------------------------------------------------------------------------
// Fragment loaders (wave32 WMMA layouts, ISA 7.12.2)
// A-matrix 16x32 bf16: lane L -> row (L&15); VGPR0-3: K = kh*8..+7, VGPR4-7: K = 16+kh*8..+7
// B-matrix 32x16 bf16 (loaded as Bt rows [N,K]): lane L -> col (L&15); VGPR0-3: K = kh*8..+7, VGPR4-7: K = 16+kh*8..+7
// ---------------------------------------------------------------------------
__device__ __forceinline__ v16bf frag_a(const __bf16* base, int ld) {
  int lane = threadIdx.x & 31;
  const __bf16* p = base + (size_t)(lane & 15) * ld + ((lane >> 4) & 1) * 8;
  v16bf f;
  reinterpret_cast<float4*>(&f)[0] = *reinterpret_cast<const float4*>(p);
  reinterpret_cast<float4*>(&f)[1] = *reinterpret_cast<const float4*>(p + 16);
  return f;
}

__device__ __forceinline__ v16bf frag_b(const __bf16* base, int ld) {
  int lane = threadIdx.x & 31;
  const __bf16* p = base + (size_t)(lane & 15) * ld + ((lane >> 4) & 1) * 8;
  v16bf f;
  reinterpret_cast<float4*>(&f)[0] = *reinterpret_cast<const float4*>(p);
  reinterpret_cast<float4*>(&f)[1] = *reinterpret_cast<const float4*>(p + 16);
  return f;
}

#if USE_TDM
// ---------------------------------------------------------------------------
// Tensor Data Mover: DMA one [rows x cols] bf16 tile (row stride ld_elems in
// global) into LDS at lds_off.  Descriptor per ISA 8.3/8.4: group0 = count=1,
// lds_addr, 57b global addr, type=2; group1 = data_size=2B, tensor dims,
// tile_dim0=cols, tile_dim1=rows, tensor_dim0_stride=ld_elems.
// ---------------------------------------------------------------------------
__device__ __forceinline__ void tdm_load_tile(const __bf16* gsrc, unsigned lds_off,
                                              int rows, int cols, int ld_elems) {
  unsigned long long ga = (unsigned long long)(size_t)gsrc;
  uint32x4 g0;
  g0[0] = 1u;                                    // count=1 (valid user descriptor)
  g0[1] = lds_off;                               // LDS byte address
  g0[2] = (unsigned)(ga & 0xFFFFFFFFu);
  g0[3] = (unsigned)((ga >> 32) & 0x01FFFFFFu) | (2u << 30);  // addr[56:32], type=2
  const unsigned TD = 1u << 20;                  // generous tensor dims (tiles in-bounds)
  int32x8 g1;
  g1[0] = (int)(1u << 16);                       // wg_mask=0, data_size=1 (2 bytes)
  g1[1] = (int)((TD & 0xFFFFu) << 16);           // tensor_dim0[15:0]
  g1[2] = (int)(((TD >> 16) & 0xFFFFu) | ((TD & 0xFFFFu) << 16));  // td0[31:16] | td1[15:0]
  g1[3] = (int)(((TD >> 16) & 0xFFFFu) | ((unsigned)cols << 16));  // td1[31:16] | tile_dim0
  g1[4] = rows;                                  // tile_dim1 (tile_dim2=0)
  g1[5] = ld_elems;                              // tensor_dim0_stride[31:0]
  g1[6] = 0;                                     // stride[47:32] | tensor_dim1_stride lo
  g1[7] = 0;
  int32x4 gz4 = {0, 0, 0, 0};
  int32x8 gz8 = {0, 0, 0, 0, 0, 0, 0, 0};
  __builtin_amdgcn_tensor_load_to_lds(g0, g1, gz4, gz4, gz8, 0);
}
#endif

// ---------------------------------------------------------------------------
// Weight transpose + cast: W[K][N] f32 -> Wt[N][K] bf16
// ---------------------------------------------------------------------------
__global__ __launch_bounds__(256) void wt_cast_kernel(const float* __restrict__ W,
                                                      __bf16* __restrict__ Wt,
                                                      int Kd, int Nd) {
  __shared__ float s[32][33];
  int tid = threadIdx.x;
  int k0 = blockIdx.y * 32, n0 = blockIdx.x * 32;
  for (int i = tid; i < 1024; i += 256) {
    int r = i >> 5, c = i & 31;
    s[r][c] = W[(size_t)(k0 + r) * Nd + n0 + c];
  }
  __syncthreads();
  for (int i = tid; i < 1024; i += 256) {
    int r = i >> 5, c = i & 31;
    Wt[(size_t)(n0 + r) * Kd + k0 + c] = (__bf16)s[c][r];
  }
}

// ---------------------------------------------------------------------------
// LayerNorm (D=1024) + cast to bf16.  One row per block, 4 floats per thread.
// ---------------------------------------------------------------------------
__global__ __launch_bounds__(256) void ln_cast_kernel(const float* __restrict__ X,
                                                      const float* __restrict__ gamma,
                                                      const float* __restrict__ beta,
                                                      __bf16* __restrict__ out) {
  int row = blockIdx.x, tid = threadIdx.x;
  int lane = tid & 31, wave = tid >> 5;
  const float* xr = X + (size_t)row * 1024;
  float4 v = reinterpret_cast<const float4*>(xr)[tid];
  float s = v.x + v.y + v.z + v.w;
  float q = v.x * v.x + v.y * v.y + v.z * v.z + v.w * v.w;
  for (int d = 1; d < 32; d <<= 1) {
    s += __shfl_xor(s, d, 32);
    q += __shfl_xor(q, d, 32);
  }
  __shared__ float ssum[8], ssq[8];
  if (lane == 0) { ssum[wave] = s; ssq[wave] = q; }
  __syncthreads();
  float tot = 0.f, totq = 0.f;
#pragma unroll
  for (int i = 0; i < 8; ++i) { tot += ssum[i]; totq += ssq[i]; }
  float mu  = tot * (1.f / 1024.f);
  float var = totq * (1.f / 1024.f) - mu * mu;
  float rs  = rsqrtf(var + 1e-5f);
  float4 g = reinterpret_cast<const float4*>(gamma)[tid];
  float4 b = reinterpret_cast<const float4*>(beta)[tid];
  v4bf o;
  o[0] = (__bf16)((v.x - mu) * rs * g.x + b.x);
  o[1] = (__bf16)((v.y - mu) * rs * g.y + b.y);
  o[2] = (__bf16)((v.z - mu) * rs * g.z + b.z);
  o[3] = (__bf16)((v.w - mu) * rs * g.w + b.w);
  reinterpret_cast<v4bf*>(out + (size_t)row * 1024)[tid] = o;
}

// ---------------------------------------------------------------------------
// Elementwise f32 -> bf16 cast (float4 granularity)
// ---------------------------------------------------------------------------
__global__ __launch_bounds__(256) void cast_bf16_kernel(const float* __restrict__ X,
                                                        __bf16* __restrict__ out, int n4) {
  int i = blockIdx.x * 256 + threadIdx.x;
  if (i < n4) {
    float4 v = reinterpret_cast<const float4*>(X)[i];
    v4bf o;
    o[0] = (__bf16)v.x; o[1] = (__bf16)v.y; o[2] = (__bf16)v.z; o[3] = (__bf16)v.w;
    reinterpret_cast<v4bf*>(out)[i] = o;
  }
}

// ---------------------------------------------------------------------------
// Tiled bf16 GEMM with WMMA.  C[M,N] = A[M,K] @ B[K,N], Bt = B^T given [N,K].
// Tile 128x128x64, 256 threads = 8 waves (4x2), each wave 32x64 of C.
// Double-buffered LDS; tiles DMA'd by the Tensor Data Mover (wave 0 issues,
// TENSORcnt-pipelined) when available.
// ---------------------------------------------------------------------------
enum {
  EPI_F32_RES = 0,       // out f32 = Res + acc
  EPI_F32_RES_BIAS = 1,  // out f32 = Res + acc + bias[n]
  EPI_BF16_SPLIT = 2,    // out bf16 [B,H,T,HD], acc*scale
  EPI_BF16_SPLIT_T = 3,  // out bf16 [B,H,HD,T], acc*scale
  EPI_BF16_GELU = 4      // out bf16 = gelu(acc + bias[n])
};

__global__ __launch_bounds__(256) void gemm_bf16_kernel(
    const __bf16* __restrict__ A, const __bf16* __restrict__ Bt,
    void* __restrict__ Cout, const float* __restrict__ Res,
    const float* __restrict__ bias, int M, int N, int K, int epi, float scale) {
  __shared__ __bf16 sA[2][128 * 64];
  __shared__ __bf16 sB[2][128 * 64];
  int tid = threadIdx.x;
  int lane = tid & 31, wave = tid >> 5;
  int wrow = wave & 3, wcol = wave >> 2;  // 4 row-groups x 2 col-groups
  int bm = blockIdx.y * 128, bn = blockIdx.x * 128;

  v8f acc[2][4] = {{{}, {}, {}, {}}, {{}, {}, {}, {}}};
  const int nk = K >> 6;

#if USE_TDM
  if (wave == 0) {
    tdm_load_tile(A + (size_t)bm * K, (unsigned)(size_t)&sA[0][0], 128, 64, K);
    tdm_load_tile(Bt + (size_t)bn * K, (unsigned)(size_t)&sB[0][0], 128, 64, K);
  }
#endif

  for (int kt = 0; kt < nk; ++kt) {
    int cur = kt & 1;
#if USE_TDM
    if (wave == 0) {
      if (kt + 1 < nk) {
        int k1 = (kt + 1) << 6;
        tdm_load_tile(A + (size_t)bm * K + k1, (unsigned)(size_t)&sA[cur ^ 1][0], 128, 64, K);
        tdm_load_tile(Bt + (size_t)bn * K + k1, (unsigned)(size_t)&sB[cur ^ 1][0], 128, 64, K);
        __builtin_amdgcn_s_wait_tensorcnt(2);  // stage kt complete (TDM in-order)
      } else {
        __builtin_amdgcn_s_wait_tensorcnt(0);
      }
    }
    __syncthreads();
#else
    {
      int k0 = kt << 6;
      int r = tid >> 3, c = (tid & 7) << 3;
      const __bf16* ga = A + (size_t)(bm + r) * K + k0 + c;
      const __bf16* gb = Bt + (size_t)(bn + r) * K + k0 + c;
      __bf16* la = &sA[cur][r * 64 + c];
      __bf16* lb = &sB[cur][r * 64 + c];
#pragma unroll
      for (int rr = 0; rr < 128; rr += 32) {
        *reinterpret_cast<float4*>(la + rr * 64) =
            *reinterpret_cast<const float4*>(ga + (size_t)rr * K);
        *reinterpret_cast<float4*>(lb + rr * 64) =
            *reinterpret_cast<const float4*>(gb + (size_t)rr * K);
      }
      __syncthreads();
    }
#endif
    const __bf16* bufA = &sA[cur][0];
    const __bf16* bufB = &sB[cur][0];
#pragma unroll
    for (int ks = 0; ks < 64; ks += 32) {
      v16bf af0 = frag_a(bufA + (wrow * 32) * 64 + ks, 64);
      v16bf af1 = frag_a(bufA + (wrow * 32 + 16) * 64 + ks, 64);
      v16bf b0 = frag_b(bufB + (wcol * 64) * 64 + ks, 64);
      v16bf b1 = frag_b(bufB + (wcol * 64 + 16) * 64 + ks, 64);
      v16bf b2 = frag_b(bufB + (wcol * 64 + 32) * 64 + ks, 64);
      v16bf b3 = frag_b(bufB + (wcol * 64 + 48) * 64 + ks, 64);
      acc[0][0] = WMMA_BF16(af0, b0, acc[0][0]);
      acc[0][1] = WMMA_BF16(af0, b1, acc[0][1]);
      acc[0][2] = WMMA_BF16(af0, b2, acc[0][2]);
      acc[0][3] = WMMA_BF16(af0, b3, acc[0][3]);
      acc[1][0] = WMMA_BF16(af1, b0, acc[1][0]);
      acc[1][1] = WMMA_BF16(af1, b1, acc[1][1]);
      acc[1][2] = WMMA_BF16(af1, b2, acc[1][2]);
      acc[1][3] = WMMA_BF16(af1, b3, acc[1][3]);
    }
    __syncthreads();
  }

  int half = (lane >> 4) & 1, n16 = lane & 15;
#pragma unroll
  for (int i = 0; i < 2; ++i)
#pragma unroll
    for (int j = 0; j < 4; ++j)
#pragma unroll
      for (int e = 0; e < 8; ++e) {
        int m = bm + wrow * 32 + i * 16 + e + half * 8;
        int n = bn + wcol * 64 + j * 16 + n16;
        float v = acc[i][j][e] * scale;
        if (epi == EPI_F32_RES) {
          ((float*)Cout)[(size_t)m * N + n] = Res[(size_t)m * N + n] + v;
        } else if (epi == EPI_F32_RES_BIAS) {
          ((float*)Cout)[(size_t)m * N + n] = Res[(size_t)m * N + n] + v + bias[n];
        } else if (epi == EPI_BF16_SPLIT) {
          int b = m >> 10, t = m & 1023, h = n >> 6, hd = n & 63;
          ((__bf16*)Cout)[(((size_t)b * 16 + h) * 1024 + t) * 64 + hd] = (__bf16)v;
        } else if (epi == EPI_BF16_SPLIT_T) {
          int b = m >> 10, t = m & 1023, h = n >> 6, hd = n & 63;
          ((__bf16*)Cout)[(((size_t)b * 16 + h) * 64 + hd) * 1024 + t] = (__bf16)v;
        } else {  // EPI_BF16_GELU
          float u = v + bias[n];
          float gv = 0.5f * u * (1.f + erff(u * 0.70710678118654752f));
          ((__bf16*)Cout)[(size_t)m * N + n] = (__bf16)gv;
        }
      }
}

// ---------------------------------------------------------------------------
// Flash attention, causal self-attention.
// Q [B,H,T,64] (pre-scaled), K [B,H,T,64], Vt [B,H,64,T], Out bf16 [B*T, 1024]
// ---------------------------------------------------------------------------
__global__ __launch_bounds__(256) void attn_self_kernel(
    const __bf16* __restrict__ Q, const __bf16* __restrict__ Kp,
    const __bf16* __restrict__ Vt, __bf16* __restrict__ Out) {
  __shared__ __bf16 sP[8][16 * 32];
  int lane = threadIdx.x & 31, wave = threadIdx.x >> 5;
  int bh = blockIdx.y;
  int b = bh >> 4, h = bh & 15;
  int q0 = blockIdx.x * 128 + wave * 16;
  int half = (lane >> 4) & 1, n16 = lane & 15;

  const __bf16* Qb = Q + ((size_t)bh * 1024 + q0) * 64;
  v16bf qf0 = frag_a(Qb, 64);
  v16bf qf1 = frag_a(Qb + 32, 64);
  v8f o0 = {}, o1 = {}, o2 = {}, o3 = {};
  float mrow[8], lrow[8];
#pragma unroll
  for (int e = 0; e < 8; ++e) { mrow[e] = NEGINF; lrow[e] = 0.f; }
  __bf16* myP = sP[wave];

  int smax = q0 + 16;
  for (int s0 = 0; s0 < smax; s0 += 32) {
    v8f lg0 = {}, lg1 = {};
    {
      const __bf16* Kb0 = Kp + ((size_t)bh * 1024 + s0) * 64;
      const __bf16* Kb1 = Kb0 + 16 * 64;
      lg0 = WMMA_BF16(qf0, frag_b(Kb0, 64), lg0);
      lg0 = WMMA_BF16(qf1, frag_b(Kb0 + 32, 64), lg0);
      lg1 = WMMA_BF16(qf0, frag_b(Kb1, 64), lg1);
      lg1 = WMMA_BF16(qf1, frag_b(Kb1 + 32, 64), lg1);
    }
#pragma unroll
    for (int e = 0; e < 8; ++e) {
      int qrow = q0 + e + half * 8;
      float a0 = lg0[e]; if (s0 + n16 > qrow) a0 = NEGINF;
      float a1 = lg1[e]; if (s0 + 16 + n16 > qrow) a1 = NEGINF;
      float mx = fmaxf(a0, a1);
      for (int d = 1; d < 16; d <<= 1) mx = fmaxf(mx, __shfl_xor(mx, d, 32));
      float mn = fmaxf(mrow[e], mx);
      float fac = (mn == mrow[e]) ? 1.f : __expf(mrow[e] - mn);
      float p0 = (a0 == NEGINF) ? 0.f : __expf(a0 - mn);
      float p1 = (a1 == NEGINF) ? 0.f : __expf(a1 - mn);
      float ps = p0 + p1;
      for (int d = 1; d < 16; d <<= 1) ps += __shfl_xor(ps, d, 32);
      lrow[e] = lrow[e] * fac + ps;
      mrow[e] = mn;
      o0[e] = o0[e] * fac; o1[e] = o1[e] * fac;
      o2[e] = o2[e] * fac; o3[e] = o3[e] * fac;
      int pr = (e + half * 8) * 32;
      myP[pr + n16] = (__bf16)p0;
      myP[pr + 16 + n16] = (__bf16)p1;
    }
    v16bf pf = frag_a(myP, 32);
    const __bf16* Vb = Vt + (size_t)bh * 64 * 1024 + s0;
    o0 = WMMA_BF16(pf, frag_b(Vb, 1024), o0);
    o1 = WMMA_BF16(pf, frag_b(Vb + 16 * 1024, 1024), o1);
    o2 = WMMA_BF16(pf, frag_b(Vb + 32 * 1024, 1024), o2);
    o3 = WMMA_BF16(pf, frag_b(Vb + 48 * 1024, 1024), o3);
  }
#pragma unroll
  for (int e = 0; e < 8; ++e) {
    float inv = lrow[e] > 0.f ? 1.f / lrow[e] : 0.f;
    int t = q0 + e + half * 8;
    size_t ro = ((size_t)b * 1024 + t) * 1024 + (size_t)h * 64 + n16;
    Out[ro]      = (__bf16)(o0[e] * inv);
    Out[ro + 16] = (__bf16)(o1[e] * inv);
    Out[ro + 32] = (__bf16)(o2[e] * inv);
    Out[ro + 48] = (__bf16)(o3[e] * inv);
  }
}

// ---------------------------------------------------------------------------
// Flash attention, cross-attention with additive bias + key padding mask.
// bias [B,1,T,S] f32, keep [B,S] int32.  Fully-masked rows output 0 (nan_to_num).
// ---------------------------------------------------------------------------
__global__ __launch_bounds__(256) void attn_cross_kernel(
    const __bf16* __restrict__ Q, const __bf16* __restrict__ Kp,
    const __bf16* __restrict__ Vt, const float* __restrict__ bias,
    const int* __restrict__ keep, __bf16* __restrict__ Out) {
  __shared__ __bf16 sP[8][16 * 32];
  int lane = threadIdx.x & 31, wave = threadIdx.x >> 5;
  int bh = blockIdx.y;
  int b = bh >> 4, h = bh & 15;
  int q0 = blockIdx.x * 128 + wave * 16;
  int half = (lane >> 4) & 1, n16 = lane & 15;

  const __bf16* Qb = Q + ((size_t)bh * 1024 + q0) * 64;
  v16bf qf0 = frag_a(Qb, 64);
  v16bf qf1 = frag_a(Qb + 32, 64);
  v8f o0 = {}, o1 = {}, o2 = {}, o3 = {};
  float mrow[8], lrow[8];
#pragma unroll
  for (int e = 0; e < 8; ++e) { mrow[e] = NEGINF; lrow[e] = 0.f; }
  __bf16* myP = sP[wave];

  for (int s0 = 0; s0 < 1024; s0 += 32) {
    v8f lg0 = {}, lg1 = {};
    {
      const __bf16* Kb0 = Kp + ((size_t)bh * 1024 + s0) * 64;
      const __bf16* Kb1 = Kb0 + 16 * 64;
      lg0 = WMMA_BF16(qf0, frag_b(Kb0, 64), lg0);
      lg0 = WMMA_BF16(qf1, frag_b(Kb0 + 32, 64), lg0);
      lg1 = WMMA_BF16(qf0, frag_b(Kb1, 64), lg1);
      lg1 = WMMA_BF16(qf1, frag_b(Kb1 + 32, 64), lg1);
    }
    int k0v = keep[b * 1024 + s0 + n16];
    int k1v = keep[b * 1024 + s0 + 16 + n16];
#pragma unroll
    for (int e = 0; e < 8; ++e) {
      int qrow = q0 + e + half * 8;
      const float* brow = bias + ((size_t)b * 1024 + qrow) * 1024 + s0 + n16;
      float a0 = lg0[e] + brow[0];  if (k0v == 0) a0 = NEGINF;
      float a1 = lg1[e] + brow[16]; if (k1v == 0) a1 = NEGINF;
      float mx = fmaxf(a0, a1);
      for (int d = 1; d < 16; d <<= 1) mx = fmaxf(mx, __shfl_xor(mx, d, 32));
      float mn = fmaxf(mrow[e], mx);
      float fac = (mn == mrow[e]) ? 1.f : __expf(mrow[e] - mn);
      float p0 = (a0 == NEGINF) ? 0.f : __expf(a0 - mn);
      float p1 = (a1 == NEGINF) ? 0.f : __expf(a1 - mn);
      float ps = p0 + p1;
      for (int d = 1; d < 16; d <<= 1) ps += __shfl_xor(ps, d, 32);
      lrow[e] = lrow[e] * fac + ps;
      mrow[e] = mn;
      o0[e] = o0[e] * fac; o1[e] = o1[e] * fac;
      o2[e] = o2[e] * fac; o3[e] = o3[e] * fac;
      int pr = (e + half * 8) * 32;
      myP[pr + n16] = (__bf16)p0;
      myP[pr + 16 + n16] = (__bf16)p1;
    }
    v16bf pf = frag_a(myP, 32);
    const __bf16* Vb = Vt + (size_t)bh * 64 * 1024 + s0;
    o0 = WMMA_BF16(pf, frag_b(Vb, 1024), o0);
    o1 = WMMA_BF16(pf, frag_b(Vb + 16 * 1024, 1024), o1);
    o2 = WMMA_BF16(pf, frag_b(Vb + 32 * 1024, 1024), o2);
    o3 = WMMA_BF16(pf, frag_b(Vb + 48 * 1024, 1024), o3);
  }
#pragma unroll
  for (int e = 0; e < 8; ++e) {
    float inv = lrow[e] > 0.f ? 1.f / lrow[e] : 0.f;  // fully-masked row -> 0
    int t = q0 + e + half * 8;
    size_t ro = ((size_t)b * 1024 + t) * 1024 + (size_t)h * 64 + n16;
    Out[ro]      = (__bf16)(o0[e] * inv);
    Out[ro + 16] = (__bf16)(o1[e] * inv);
    Out[ro + 32] = (__bf16)(o2[e] * inv);
    Out[ro + 48] = (__bf16)(o3[e] * inv);
  }
}

// ---------------------------------------------------------------------------
// Host orchestration
// ---------------------------------------------------------------------------
extern "C" void kernel_launch(void* const* d_in, const int* in_sizes, int n_in,
                              void* d_out, int out_size, void* d_ws, size_t ws_size,
                              hipStream_t stream) {
  (void)in_sizes; (void)n_in; (void)out_size; (void)ws_size;
  const int D = 1024, DFF = 4096, BT = 4096, BS = 4096;

  const float* x     = (const float*)d_in[0];
  const float* audio = (const float*)d_in[1];
  const float* bias  = (const float*)d_in[2];
  const int*   kpm   = (const int*)d_in[3];
  const float* ln1g = (const float*)d_in[4],  *ln1b = (const float*)d_in[5];
  const float* ln2g = (const float*)d_in[6],  *ln2b = (const float*)d_in[7];
  const float* ln3g = (const float*)d_in[8],  *ln3b = (const float*)d_in[9];
  const float* W[10] = {(const float*)d_in[10], (const float*)d_in[11],
                        (const float*)d_in[12], (const float*)d_in[13],
                        (const float*)d_in[14], (const float*)d_in[15],
                        (const float*)d_in[16], (const float*)d_in[17],
                        (const float*)d_in[18], (const float*)d_in[20]};
  const float* bf1v = (const float*)d_in[19];
  const float* bf2v = (const float*)d_in[21];

  char* w = (char*)d_ws;
  size_t off = 0;
  auto take = [&](size_t bytes) -> void* {
    void* p = w + off;
    off = (off + bytes + 255) & ~(size_t)255;
    return p;
  };
  __bf16* Wt[10];
  for (int i = 0; i < 8; ++i) Wt[i] = (__bf16*)take((size_t)D * D * 2);
  Wt[8] = (__bf16*)take((size_t)D * DFF * 2);  // Wf1^T [DFF, D]
  Wt[9] = (__bf16*)take((size_t)D * DFF * 2);  // Wf2^T [D, DFF]
  __bf16* hb   = (__bf16*)take((size_t)BT * D * 2);
  __bf16* acb  = (__bf16*)take((size_t)BS * D * 2);
  __bf16* Qb   = (__bf16*)take((size_t)BT * D * 2);
  __bf16* Kb   = (__bf16*)take((size_t)BT * D * 2);
  __bf16* Vtb  = (__bf16*)take((size_t)BT * D * 2);
  __bf16* aout = (__bf16*)take((size_t)BT * D * 2);
  float*  x1   = (float*)take((size_t)BT * D * 4);
  float*  x2   = (float*)take((size_t)BT * D * 4);
  __bf16* ffb  = (__bf16*)take((size_t)BT * DFF * 2);

  // Weight transpose+cast: Wq..Wco are [D,D]; Wf1 is [D,DFF]; Wf2 is [DFF,D]
  for (int i = 0; i < 8; ++i)
    wt_cast_kernel<<<dim3(D / 32, D / 32), 256, 0, stream>>>(W[i], Wt[i], D, D);
  wt_cast_kernel<<<dim3(DFF / 32, D / 32), 256, 0, stream>>>(W[8], Wt[8], D, DFF);
  wt_cast_kernel<<<dim3(D / 32, DFF / 32), 256, 0, stream>>>(W[9], Wt[9], DFF, D);

  auto gemm = [&](const __bf16* A, const __bf16* Bt, void* C, const float* Res,
                  const float* bb, int M, int N, int K, int epi, float scale) {
    gemm_bf16_kernel<<<dim3(N / 128, M / 128), 256, 0, stream>>>(A, Bt, C, Res, bb,
                                                                 M, N, K, epi, scale);
  };
  const float scl = 0.125f;  // 1/sqrt(64)

  // --- causal self-attention ---
  ln_cast_kernel<<<BT, 256, 0, stream>>>(x, ln1g, ln1b, hb);
  gemm(hb, Wt[0], Qb,  nullptr, nullptr, BT, D, D, EPI_BF16_SPLIT,   scl);
  gemm(hb, Wt[1], Kb,  nullptr, nullptr, BT, D, D, EPI_BF16_SPLIT,   1.f);
  gemm(hb, Wt[2], Vtb, nullptr, nullptr, BT, D, D, EPI_BF16_SPLIT_T, 1.f);
  attn_self_kernel<<<dim3(8, 64), 256, 0, stream>>>(Qb, Kb, Vtb, aout);
  gemm(aout, Wt[3], x1, x, nullptr, BT, D, D, EPI_F32_RES, 1.f);

  // --- biased masked cross-attention ---
  ln_cast_kernel<<<BT, 256, 0, stream>>>(x1, ln2g, ln2b, hb);
  cast_bf16_kernel<<<(BS * D / 4 + 255) / 256, 256, 0, stream>>>(audio, acb, BS * D / 4);
  gemm(hb,  Wt[4], Qb,  nullptr, nullptr, BT, D, D, EPI_BF16_SPLIT,   scl);
  gemm(acb, Wt[5], Kb,  nullptr, nullptr, BS, D, D, EPI_BF16_SPLIT,   1.f);
  gemm(acb, Wt[6], Vtb, nullptr, nullptr, BS, D, D, EPI_BF16_SPLIT_T, 1.f);
  attn_cross_kernel<<<dim3(8, 64), 256, 0, stream>>>(Qb, Kb, Vtb, bias, kpm, aout);
  gemm(aout, Wt[7], x2, x1, nullptr, BT, D, D, EPI_F32_RES, 1.f);

  // --- FFN ---
  ln_cast_kernel<<<BT, 256, 0, stream>>>(x2, ln3g, ln3b, hb);
  gemm(hb,  Wt[8], ffb,   nullptr, bf1v, BT, DFF, D, EPI_BF16_GELU, 1.f);
  gemm(ffb, Wt[9], d_out, x2,      bf2v, BT, D, DFF, EPI_F32_RES_BIAS, 1.f);
}
